// MAGAT_9509057593833
// MI455X (gfx1250) — matrix operand order repaired and checked by reading, not compiled, not run on hardware
//
#include <hip/hip_runtime.h>
#include <math.h>

// ---------------------------------------------------------------------------
// MAGAT pipeline for MI455X (gfx1250, wave32). All GEMM-shaped work (asset MLP,
// GAT projections, and the GAT attention aggregation) on v_wmma_f32_16x16x32_f16.
// B=32, N=512, L=60, F=16, GH=64, H=4, MH=32, PH=128, V=512
// ---------------------------------------------------------------------------

typedef __attribute__((ext_vector_type(16))) _Float16 v16h;
typedef __attribute__((ext_vector_type(8)))  _Float16 v8h;
typedef __attribute__((ext_vector_type(8)))  float    v8f;

// ---- WMMA fragment helpers (contiguous K order, same for A and B) ------------------
// A (16x32 f16), LDS row-major, `stride` halves between rows.
// lane<16 : row=lane,    K = 0..15
// lane>=16: row=lane-16, K = 16..31
static __device__ inline v16h load_a_frag(const _Float16* tile, int stride, int lane) {
    const int r  = lane & 15;
    const int hi = lane >> 4;
    union { v16h v; v8h h[2]; } u;
    const _Float16* rp = tile + r * stride;
    u.h[0] = *(const v8h*)(rp + hi * 16);
    u.h[1] = *(const v8h*)(rp + hi * 16 + 8);
    return u.v;
}

// B (32x16 f16) stored TRANSPOSED in LDS: bt[n][k], `stride` halves per n-row.
// lane<16 : col=lane,    K = 0..15 (VGPR v holds K=2v,2v+1)
// lane>=16: col=lane-16, K = 16..31
static __device__ inline v16h load_bt_frag(const _Float16* bt, int stride, int lane) {
    const int n  = lane & 15;
    const int hi = lane >> 4;
    return *(const v16h*)(bt + n * stride + hi * 16);
}

static __device__ inline v8f wmma_f16(v16h a, v16h b, v8f c) {
    return __builtin_amdgcn_wmma_f32_16x16x32_f16(false, a, false, b, (short)0, c,
                                                  false, false);
}

static __device__ inline float gelu_f(float x) {
    return 0.5f * x * (1.0f + erff(x * 0.70710678118654752f));
}

// ---------------------------------------------------------------------------
// K1: asset encoder.  out = LN(gelu(gelu(x@W1+b1)@W2+b2))
// grid.x = (B*N)/64 = 256 blocks, 256 threads (8 waves).
// ---------------------------------------------------------------------------
__global__ __launch_bounds__(256) void k_asset_enc(
    const float* __restrict__ x,  const float* __restrict__ w1, const float* __restrict__ b1,
    const float* __restrict__ w2, const float* __restrict__ b2,
    const float* __restrict__ g,  const float* __restrict__ beta,
    float* __restrict__ out)
{
    __shared__ __align__(32) _Float16 As[64 * 32];   // A staging (f32->f16)
    __shared__ __align__(32) _Float16 Bs[128 * 32];  // transposed weight staging
    __shared__ __align__(32) _Float16 Hs[64 * 128];  // hidden activations (f16)
    __shared__ float Os[64 * 64];                    // pre-LN output

    const int tid  = threadIdx.x;
    const int lane = tid & 31;
    const int w    = tid >> 5;
    const int r0   = blockIdx.x * 64;
    const int rt   = w & 3;              // 16-row tile owned by this wave

    // ---- phase 1: hidden(64x128) = gelu(x(64x960) @ w1(960x128) + b1)
    {
        const int cg = w >> 2;           // 2 groups of 4 col-tiles
        v8f acc[4];
        for (int q = 0; q < 4; ++q) for (int r = 0; r < 8; ++r) acc[q][r] = 0.f;
        for (int kk = 0; kk < 960; kk += 32) {
            for (int i = tid; i < 64 * 32; i += 256)
                As[i] = (_Float16)x[(size_t)(r0 + (i >> 5)) * 960 + kk + (i & 31)];
            for (int i = tid; i < 128 * 32; i += 256)
                Bs[i] = (_Float16)w1[(size_t)(kk + (i & 31)) * 128 + (i >> 5)];
            __syncthreads();
            v16h a = load_a_frag(As + rt * 16 * 32, 32, lane);
            for (int q = 0; q < 4; ++q) {
                v16h b = load_bt_frag(Bs + (cg * 4 + q) * 16 * 32, 32, lane);
                acc[q] = wmma_f16(a, b, acc[q]);
            }
            __syncthreads();
        }
        const int col0 = lane & 15, hi = lane >> 4;
        for (int q = 0; q < 4; ++q) {
            const int c = (cg * 4 + q) * 16 + col0;
            const float bb = b1[c];
            for (int r = 0; r < 8; ++r) {
                const int row = rt * 16 + r + hi * 8;
                Hs[row * 128 + c] = (_Float16)gelu_f(acc[q][r] + bb);
            }
        }
    }
    __syncthreads();

    // ---- phase 2: out(64x64) = gelu(hidden @ w2(128x64) + b2)
    {
        const int ct0 = (w >> 2) * 2;
        v8f acc[2];
        for (int q = 0; q < 2; ++q) for (int r = 0; r < 8; ++r) acc[q][r] = 0.f;
        for (int kk = 0; kk < 128; kk += 32) {
            for (int i = tid; i < 64 * 32; i += 256)
                Bs[i] = (_Float16)w2[(size_t)(kk + (i & 31)) * 64 + (i >> 5)];
            __syncthreads();
            v16h a = load_a_frag(Hs + rt * 16 * 128 + kk, 128, lane);
            for (int q = 0; q < 2; ++q) {
                v16h b = load_bt_frag(Bs + (ct0 + q) * 16 * 32, 32, lane);
                acc[q] = wmma_f16(a, b, acc[q]);
            }
            __syncthreads();
        }
        const int col0 = lane & 15, hi = lane >> 4;
        for (int q = 0; q < 2; ++q) {
            const int c = (ct0 + q) * 16 + col0;
            const float bb = b2[c];
            for (int r = 0; r < 8; ++r) {
                const int row = rt * 16 + r + hi * 8;
                Os[row * 64 + c] = gelu_f(acc[q][r] + bb);
            }
        }
    }
    __syncthreads();

    // ---- LayerNorm per row (64 elems)
    if (tid < 64) {
        float m = 0.f;
        for (int c = 0; c < 64; ++c) m += Os[tid * 64 + c];
        m *= (1.f / 64.f);
        float v = 0.f;
        for (int c = 0; c < 64; ++c) { float d = Os[tid * 64 + c] - m; v += d * d; }
        v *= (1.f / 64.f);
        const float inv = rsqrtf(v + 1e-5f);
        for (int c = 0; c < 64; ++c)
            out[(size_t)(r0 + tid) * 64 + c] = (Os[tid * 64 + c] - m) * inv * g[c] + beta[c];
    }
}

// ---------------------------------------------------------------------------
// K2: macro encoder. ctx(B,32) = LN(mean_l(gelu(x@w1+b1)@w2+b2))
// ---------------------------------------------------------------------------
__global__ __launch_bounds__(256) void k_macro(
    const float* __restrict__ xm,
    const float* __restrict__ w1, const float* __restrict__ b1,
    const float* __restrict__ w2, const float* __restrict__ b2,
    const float* __restrict__ g,  const float* __restrict__ bt,
    float* __restrict__ ctx)
{
    __shared__ float h[60 * 32];
    __shared__ float ms[32];
    const int b = blockIdx.x, tid = threadIdx.x;
    for (int i = tid; i < 60 * 32; i += 256) {
        const int l = i >> 5, j = i & 31;
        float a = b1[j];
        const float* xr = &xm[((size_t)b * 60 + l) * 32];
        for (int k = 0; k < 32; ++k) a += xr[k] * w1[k * 32 + j];
        h[i] = gelu_f(a);
    }
    __syncthreads();
    if (tid < 32) {
        float mm = 0.f;
        for (int l = 0; l < 60; ++l) {
            float a = b2[tid];
            for (int k = 0; k < 32; ++k) a += h[l * 32 + k] * w2[k * 32 + tid];
            mm += a;
        }
        ms[tid] = mm * (1.f / 60.f);
    }
    __syncthreads();
    if (tid < 32) {
        float m = 0.f;
        for (int j = 0; j < 32; ++j) m += ms[j];
        m *= (1.f / 32.f);
        float v = 0.f;
        for (int j = 0; j < 32; ++j) { float d = ms[j] - m; v += d * d; }
        v *= (1.f / 32.f);
        ctx[(size_t)b * 32 + tid] = (ms[tid] - m) * rsqrtf(v + 1e-5f) * g[tid] + bt[tid];
    }
}

// ---------------------------------------------------------------------------
// K3/K5: GAT projection  Wx = X(16384x64) @ W(64xOUTW), fused s/t dot epilogue.
// grid: (256, OUTW/64) blocks; each block = 64 rows x 64 cols.
// D = head dim (16 or 64); OUTW = H*D.
// ---------------------------------------------------------------------------
template <int OUTW, int D>
__global__ __launch_bounds__(256) void k_gat_wx(
    const float* __restrict__ X, const float* __restrict__ W, const float* __restrict__ attn,
    float* __restrict__ Wx, float* __restrict__ s, float* __restrict__ t)
{
    __shared__ __align__(32) _Float16 As[64 * 32];
    __shared__ __align__(32) _Float16 Bs[64 * 32];
    __shared__ float Os[64 * 64];
    const int tid  = threadIdx.x;
    const int lane = tid & 31;
    const int w    = tid >> 5;
    const int r0   = blockIdx.x * 64;
    const int cb   = blockIdx.y * 64;
    const int rt   = w & 3;
    const int ct0  = (w >> 2) * 2;

    v8f acc[2];
    for (int q = 0; q < 2; ++q) for (int r = 0; r < 8; ++r) acc[q][r] = 0.f;
    for (int kk = 0; kk < 64; kk += 32) {
        for (int i = tid; i < 64 * 32; i += 256)
            As[i] = (_Float16)X[(size_t)(r0 + (i >> 5)) * 64 + kk + (i & 31)];
        for (int i = tid; i < 64 * 32; i += 256)
            Bs[i] = (_Float16)W[(size_t)(kk + (i & 31)) * OUTW + cb + (i >> 5)];
        __syncthreads();
        v16h a = load_a_frag(As + rt * 16 * 32, 32, lane);
        for (int q = 0; q < 2; ++q) {
            v16h b = load_bt_frag(Bs + (ct0 + q) * 16 * 32, 32, lane);
            acc[q] = wmma_f16(a, b, acc[q]);
        }
        __syncthreads();
    }
    {
        const int col0 = lane & 15, hi = lane >> 4;
        for (int q = 0; q < 2; ++q) {
            const int c = (ct0 + q) * 16 + col0;
            for (int r = 0; r < 8; ++r) {
                const int row = rt * 16 + r + hi * 8;
                const float v = acc[q][r];
                Os[row * 64 + c] = v;
                Wx[(size_t)(r0 + row) * OUTW + cb + c] = v;
            }
        }
    }
    __syncthreads();
    // s[n,h] = Wx[n,h,:].attn[h,:D]; t with attn[h,D:2D].  Heads fully inside block.
    constexpr int HPB = 64 / D;  // heads covered by this 64-col block
    constexpr int H   = 4;
    if (tid < 64 * HPB) {
        const int row = tid / HPB, hl = tid % HPB;
        const int hg  = cb / D + hl;
        float sa = 0.f, ta = 0.f;
        for (int d = 0; d < D; ++d) {
            const float v = Os[row * 64 + hl * D + d];
            sa += v * attn[hg * 2 * D + d];
            ta += v * attn[hg * 2 * D + D + d];
        }
        s[(size_t)(r0 + row) * H + hg] = sa;
        t[(size_t)(r0 + row) * H + hg] = ta;
    }
}

// ---------------------------------------------------------------------------
// K4/K6: WMMA attention aggregation per (b,h):
//   out[i,:] = sum_j softmax_j(leaky_relu(s_i + t_j)) * Wx[j,:]   (+ optional ELU)
// Never materializes (N,N): logits are outer-sums. Per 16-row tile a VALU
// prepass computes row max & sum; the probability A-fragment is synthesized
// directly in VGPRs in contiguous K order; B-fragments are contiguous v16h loads from
// a transposed f16 Wx image in LDS; P@Wx accumulates on the WMMA pipe.
// ---------------------------------------------------------------------------
template <int D, bool ELU>
__global__ __launch_bounds__(256) void k_att_wmma(
    const float* __restrict__ Wx, const float* __restrict__ s, const float* __restrict__ t,
    float* __restrict__ out)
{
    constexpr int H = 4, N = 512, OUTW = H * D;
    constexpr int CT = D / 16;                       // 16-col output tiles
    __shared__ float tsh[N];
    __shared__ float invsh[128];                     // 8 waves x 16 rows
    __shared__ __align__(32) _Float16 WxT[D * N];    // WxT[d][j] (f16)

    const int b = blockIdx.x, h = blockIdx.y, tid = threadIdx.x;
    const int lane = tid & 31, w = tid >> 5;
    const int r = lane & 15, hi = lane >> 4;

    for (int i = tid; i < N; i += 256) tsh[i] = t[((size_t)b * N + i) * H + h];
    for (int i = tid; i < D * N; i += 256) {
        const int d = i % D, j = i / D;              // consecutive tids: contiguous global
        WxT[d * N + j] = (_Float16)Wx[((size_t)b * N + j) * OUTW + h * D + d];
    }
    __syncthreads();

    for (int rt = 0; rt < 4; ++rt) {                 // wave w owns row tiles w*4+rt
        const int i0 = (w * 4 + rt) * 16;
        const float si = s[((size_t)b * N + i0 + r) * H + h];
        // rowwise max then sum of exp (redundant across the two lane halves)
        float m = -1e30f;
        for (int j = 0; j < N; ++j) {
            float e = si + tsh[j];
            e = e > 0.f ? e : 0.2f * e;
            m = fmaxf(m, e);
        }
        float sum = 0.f;
        for (int j = 0; j < N; ++j) {
            float e = si + tsh[j];
            e = e > 0.f ? e : 0.2f * e;
            sum += expf(e - m);
        }
        if (hi == 0) invsh[w * 16 + r] = 1.f / sum;

        v8f acc[CT];
        for (int q = 0; q < CT; ++q) for (int k = 0; k < 8; ++k) acc[q][k] = 0.f;
        for (int jt = 0; jt < N / 32; ++jt) {
            // synthesize P fragment (16x32 f16) in A layout:
            // lane half hi owns K = hi*16+c and hi*16+8+c
            v16h pa;
            for (int c = 0; c < 8; ++c) {
                const int j0 = jt * 32 + hi * 16 + c;
                const int j1 = j0 + 8;
                float e0 = si + tsh[j0]; e0 = e0 > 0.f ? e0 : 0.2f * e0;
                float e1 = si + tsh[j1]; e1 = e1 > 0.f ? e1 : 0.2f * e1;
                pa[c]     = (_Float16)expf(e0 - m);
                pa[8 + c] = (_Float16)expf(e1 - m);
            }
            for (int q = 0; q < CT; ++q) {
                v16h bfr = *(const v16h*)(&WxT[(q * 16 + r) * N + jt * 32 + hi * 16]);
                acc[q] = wmma_f16(pa, bfr, acc[q]);
            }
        }
        // epilogue: normalize, optional ELU, store. C layout: rows rr+hi*8, col r.
        for (int q = 0; q < CT; ++q) {
            for (int rr = 0; rr < 8; ++rr) {
                const int rowc = rr + hi * 8;
                const float inv = invsh[w * 16 + rowc];
                float v = acc[q][rr] * inv;
                if (ELU) v = v > 0.f ? v : (expf(v) - 1.f);
                out[((size_t)b * N + i0 + rowc) * OUTW + h * D + q * 16 + r] = v;
            }
        }
    }
}

// ---------------------------------------------------------------------------
// K7: mean over heads + ELU + LayerNorm.  One thread per (b,n) row.
// ---------------------------------------------------------------------------
__global__ __launch_bounds__(256) void k_headmean_ln(
    const float* __restrict__ att, const float* __restrict__ g, const float* __restrict__ b,
    float* __restrict__ out)
{
    const int row = blockIdx.x * 256 + threadIdx.x;  // 0..16383
    const float* p = &att[(size_t)row * 256];
    float v[64];
    float m = 0.f;
    for (int d = 0; d < 64; ++d) {
        float a = 0.25f * (p[d] + p[64 + d] + p[128 + d] + p[192 + d]);
        a = a > 0.f ? a : (expf(a) - 1.f);
        v[d] = a;
        m += a;
    }
    m *= (1.f / 64.f);
    float var = 0.f;
    for (int d = 0; d < 64; ++d) { float dd = v[d] - m; var += dd * dd; }
    var *= (1.f / 64.f);
    const float inv = rsqrtf(var + 1e-5f);
    for (int d = 0; d < 64; ++d)
        out[(size_t)row * 64 + d] = (v[d] - m) * inv * g[d] + b[d];
}

// ---------------------------------------------------------------------------
// K8: mean-pool over assets.  one block per batch.
// ---------------------------------------------------------------------------
__global__ __launch_bounds__(256) void k_pool(
    const float* __restrict__ ge, float* __restrict__ pool)
{
    __shared__ float red[256];
    const int b = blockIdx.x, tid = threadIdx.x;
    const int d = tid & 63, ch = tid >> 6;
    float a = 0.f;
    for (int i = ch * 128; i < (ch + 1) * 128; ++i)
        a += ge[((size_t)b * 512 + i) * 64 + d];
    red[tid] = a;
    __syncthreads();
    if (tid < 64)
        pool[(size_t)b * 64 + tid] =
            (red[tid] + red[64 + tid] + red[128 + tid] + red[192 + tid]) * (1.f / 512.f);
}

// ---------------------------------------------------------------------------
// K9: portfolio head + softmax. one block per batch.
// ---------------------------------------------------------------------------
__global__ __launch_bounds__(256) void k_port(
    const float* __restrict__ pool, const float* __restrict__ mctx,
    const float* __restrict__ w1, const float* __restrict__ b1,
    const float* __restrict__ w2, const float* __restrict__ b2,
    const float* __restrict__ w3, const float* __restrict__ b3,
    float* __restrict__ out)
{
    __shared__ float comb[96];
    __shared__ float z1[128];
    __shared__ float z2[64];
    __shared__ float red[256];
    const int b = blockIdx.x, tid = threadIdx.x;
    if (tid < 64)      comb[tid] = pool[(size_t)b * 64 + tid];
    else if (tid < 96) comb[tid] = mctx[(size_t)b * 32 + tid - 64];
    __syncthreads();
    if (tid < 128) {
        float a = b1[tid];
        for (int k = 0; k < 96; ++k) a += comb[k] * w1[k * 128 + tid];
        z1[tid] = gelu_f(a);
    }
    __syncthreads();
    if (tid < 64) {
        float a = b2[tid];
        for (int k = 0; k < 128; ++k) a += z1[k] * w2[k * 64 + tid];
        z2[tid] = gelu_f(a);
    }
    __syncthreads();
    float l0 = b3[tid], l1 = b3[tid + 256];
    for (int k = 0; k < 64; ++k) {
        const float z = z2[k];
        l0 += z * w3[k * 512 + tid];
        l1 += z * w3[k * 512 + tid + 256];
    }
    red[tid] = fmaxf(l0, l1);
    __syncthreads();
    for (int st = 128; st > 0; st >>= 1) {
        if (tid < st) red[tid] = fmaxf(red[tid], red[tid + st]);
        __syncthreads();
    }
    const float m = red[0];
    __syncthreads();
    const float e0 = expf(l0 - m), e1 = expf(l1 - m);
    red[tid] = e0 + e1;
    __syncthreads();
    for (int st = 128; st > 0; st >>= 1) {
        if (tid < st) red[tid] += red[tid + st];
        __syncthreads();
    }
    const float inv = 1.f / red[0];
    out[(size_t)b * 512 + tid]       = e0 * inv;
    out[(size_t)b * 512 + tid + 256] = e1 * inv;
}

// ---------------------------------------------------------------------------
// launch
// ---------------------------------------------------------------------------
extern "C" void kernel_launch(void* const* d_in, const int* in_sizes, int n_in,
                              void* d_out, int out_size, void* d_ws, size_t ws_size,
                              hipStream_t stream)
{
    (void)in_sizes; (void)n_in; (void)out_size; (void)ws_size;
    const float* x_asset = (const float*)d_in[0];
    const float* x_macro = (const float*)d_in[1];
    const float* ae_w1   = (const float*)d_in[2];
    const float* ae_b1   = (const float*)d_in[3];
    const float* ae_w2   = (const float*)d_in[4];
    const float* ae_b2   = (const float*)d_in[5];
    const float* ae_g    = (const float*)d_in[6];
    const float* ae_bt   = (const float*)d_in[7];
    const float* me_w1   = (const float*)d_in[8];
    const float* me_b1   = (const float*)d_in[9];
    const float* me_w2   = (const float*)d_in[10];
    const float* me_b2   = (const float*)d_in[11];
    const float* me_g    = (const float*)d_in[12];
    const float* me_bt   = (const float*)d_in[13];
    const float* g1_w    = (const float*)d_in[14];
    const float* g1_a    = (const float*)d_in[15];
    const float* g2_w    = (const float*)d_in[16];
    const float* g2_a    = (const float*)d_in[17];
    const float* gn_g    = (const float*)d_in[18];
    const float* gn_b    = (const float*)d_in[19];
    const float* ph_w1   = (const float*)d_in[20];
    const float* ph_b1   = (const float*)d_in[21];
    const float* ph_w2   = (const float*)d_in[22];
    const float* ph_b2   = (const float*)d_in[23];
    const float* ph_w3   = (const float*)d_in[24];
    const float* ph_b3   = (const float*)d_in[25];
    float* out = (float*)d_out;

    // workspace layout (floats); reuse: bufA = asset_emb then h1,
    // bufB = Wx1 then graph_emb, s/t shared between the two GAT layers.
    float* ws   = (float*)d_ws;
    float* bufA = ws;                  // 1,048,576  asset_emb / h1
    float* bufB = ws + 1048576;        // 1,048,576  Wx1 / graph_emb
    float* Wx2  = ws + 2097152;        // 4,194,304
    float* att2 = ws + 6291456;        // 4,194,304
    float* sbuf = ws + 10485760;       // 65,536
    float* tbuf = ws + 10551296;       // 65,536
    float* mctx = ws + 10616832;       // 1,024
    float* pool = ws + 10617856;       // 2,048

    k_asset_enc<<<256, 256, 0, stream>>>(x_asset, ae_w1, ae_b1, ae_w2, ae_b2,
                                         ae_g, ae_bt, bufA);
    k_macro<<<32, 256, 0, stream>>>(x_macro, me_w1, me_b1, me_w2, me_b2,
                                    me_g, me_bt, mctx);
    // GAT layer 1 (concat=True, D=16, OUTW=64)
    k_gat_wx<64, 16><<<dim3(256, 1), 256, 0, stream>>>(bufA, g1_w, g1_a,
                                                       bufB, sbuf, tbuf);
    k_att_wmma<16, true><<<dim3(32, 4), 256, 0, stream>>>(bufB, sbuf, tbuf, bufA);
    // GAT layer 2 (concat=False, D=64, OUTW=256)
    k_gat_wx<256, 64><<<dim3(256, 4), 256, 0, stream>>>(bufA, g2_w, g2_a,
                                                        Wx2, sbuf, tbuf);
    k_att_wmma<64, false><<<dim3(32, 4), 256, 0, stream>>>(Wx2, sbuf, tbuf, att2);
    k_headmean_ln<<<64, 256, 0, stream>>>(att2, gn_g, gn_b, bufB);
    k_pool<<<32, 256, 0, stream>>>(bufB, pool);
    k_port<<<32, 256, 0, stream>>>(pool, mctx, ph_w1, ph_b1, ph_w2, ph_b2,
                                   ph_w3, ph_b3, out);
}
